// Social_Aggregator_13022340842207
// MI455X (gfx1250) — matrix operand order repaired and checked by reading, not compiled, not run on hardware
//
#include <hip/hip_runtime.h>
#include <hip/hip_bf16.h>
#include <cstdint>
#include <cstddef>

// ---------------- problem constants (match reference) ----------------
#define N_NODES 100000
#define DIMS    64

// ---------------- CDNA5 WMMA vector types ----------------
typedef __attribute__((ext_vector_type(16))) __bf16 bf16x16;
typedef __attribute__((ext_vector_type(8)))  float  floatx8;

// fp32 -> bf16 via native hardware conversion (RNE); lets clang emit
// v_cvt_pk_bf16_f32-class packed converts instead of bit-twiddling.
__device__ __forceinline__ __bf16 f2bf(float f) {
    return (__bf16)f;
}

// monotone float <-> uint mapping for atomic segment-max
__device__ __forceinline__ unsigned flipf(float f) {
    unsigned u = __builtin_bit_cast(unsigned, f);
    return (u & 0x80000000u) ? ~u : (u | 0x80000000u);
}
__device__ __forceinline__ float unflipf(unsigned u) {
    unsigned v = (u & 0x80000000u) ? (u & 0x7FFFFFFFu) : ~u;
    return __builtin_bit_cast(float, v);
}

// ---------------------------------------------------------------------
// Kernel 0: pre-swizzle W1/W2 into per-lane WMMA B-fragment layout (bf16).
// B fragment (K=32 x N=16, 16-bit): lane = half*16 + n ; element i -> K = 8*half + (i&7) + 16*(i>>3)
// w1f layout: [nt(4)][kc(4)][lane(32)][i(16)]   (K-chunk kc covers K=32*kc..)
// w2f layout: [nt(4)][kc(2)][lane(32)][i(16)]
// ---------------------------------------------------------------------
__global__ void prep_weights_kernel(const float* __restrict__ W1,
                                    const float* __restrict__ W2,
                                    __bf16* __restrict__ w1f,
                                    __bf16* __restrict__ w2f) {
    int t = blockIdx.x * blockDim.x + threadIdx.x;
    if (t < 4 * 4 * 32 * 16) {               // W1 fragments: 8192 elems
        int i    = t & 15;
        int lane = (t >> 4) & 31;
        int kc   = (t >> 9) & 3;
        int nt   = t >> 11;
        int half = lane >> 4, n = lane & 15;
        int k = 32 * kc + 8 * half + (i & 7) + 16 * (i >> 3);     // K in [0,128)
        w1f[t] = f2bf(W1[k * DIMS + nt * 16 + n]);
    }
    if (t < 4 * 2 * 32 * 16) {               // W2 fragments: 4096 elems
        int i    = t & 15;
        int lane = (t >> 4) & 31;
        int kc   = (t >> 9) & 1;
        int nt   = t >> 10;
        int half = lane >> 4, n = lane & 15;
        int k = 32 * kc + 8 * half + (i & 7) + 16 * (i >> 3);     // K in [0,64)
        w2f[t] = f2bf(W2[k * DIMS + nt * 16 + n]);
    }
}

// ---------------------------------------------------------------------
// Kernel 1: zero/identity-init segment buffers.
// ---------------------------------------------------------------------
__global__ void init_segments_kernel(unsigned* __restrict__ maxkey,
                                     float* __restrict__ segsum,
                                     float* __restrict__ agg) {
    int t = blockIdx.x * blockDim.x + threadIdx.x;
    if (t < N_NODES * DIMS) agg[t] = 0.f;
    if (t < N_NODES) { maxkey[t] = 0u; segsum[t] = 0.f; }
}

// ---------------------------------------------------------------------
// Kernel 2: per-edge MLP -> score, via bf16 WMMA (16 edges per wave).
//   layer1: [16,128]bf16 @ [128,64]bf16 + b1, relu   (16 wmma)
//   layer2: [16, 64]bf16 @ [ 64,64]bf16 + b2, relu   ( 8 wmma)
//   head:   dot with W3 + b3 via cross-lane reduction
// ---------------------------------------------------------------------
__global__ __launch_bounds__(128, 1)
void mlp_score_kernel(const float* __restrict__ embed_u,
                      const float* __restrict__ rep,
                      const __bf16* __restrict__ w1f,
                      const __bf16* __restrict__ w2f,
                      const float* __restrict__ b1,
                      const float* __restrict__ b2,
                      const float* __restrict__ W3,
                      const float* __restrict__ b3,
                      float* __restrict__ scores,
                      int E) {
    __shared__ float h1[4][16 * 68];          // 4 waves, 16x64 fp32, pitch 68 (bank-conflict pad)

    const int lane = threadIdx.x & 31;
    const int wave = threadIdx.x >> 5;
    const int tile = blockIdx.x * 4 + wave;
    if (tile * 16 + 16 > E) return;
    const int half = lane >> 4;               // lane half selects K sub-block
    const int m    = lane & 15;               // A-row / B-column within tile
    const int edge = tile * 16 + m;

    const float* rowU = embed_u + (size_t)edge * DIMS;
    const float* rowR = rep     + (size_t)edge * DIMS;

    // ---- build layer-1 A fragments (16x32 bf16, ISA A-layout) ----
    bf16x16 a1[4];
#pragma unroll
    for (int kc = 0; kc < 4; ++kc) {
        const int k0 = 32 * kc + 8 * half;    // elements 0..7  -> K = k0..k0+7
        const int k1 = k0 + 16;               // elements 8..15 -> K = k1..k1+7
        const float* p0 = (k0 < DIMS) ? (rowU + k0) : (rowR + (k0 - DIMS));
        const float* p1 = (k1 < DIMS) ? (rowU + k1) : (rowR + (k1 - DIMS));
        float tmp[16];
        *(float4*)&tmp[0]  = *(const float4*)(p0);
        *(float4*)&tmp[4]  = *(const float4*)(p0 + 4);
        *(float4*)&tmp[8]  = *(const float4*)(p1);
        *(float4*)&tmp[12] = *(const float4*)(p1 + 4);
#pragma unroll
        for (int i = 0; i < 16; ++i) a1[kc][i] = f2bf(tmp[i]);
    }

    // ---- layer 1: 4 N-tiles x 4 K-chunks of v_wmma_f32_16x16x32_bf16 ----
#pragma unroll
    for (int nt = 0; nt < 4; ++nt) {
        const float bias = b1[nt * 16 + m];   // per output column n = lane&15
        floatx8 c;
#pragma unroll
        for (int e = 0; e < 8; ++e) c[e] = bias;
#pragma unroll
        for (int kc = 0; kc < 4; ++kc) {
            const bf16x16 bw =
                *(const bf16x16*)(w1f + (size_t)((nt * 4 + kc) * 32 + lane) * 16);
            c = __builtin_amdgcn_wmma_f32_16x16x32_bf16(
                    false, a1[kc], false, bw, (short)0, c, false, false);
        }
        // relu + stash to LDS in row-major (D-layout: VGPR e -> row e + 8*half, col lane&15)
#pragma unroll
        for (int e = 0; e < 8; ++e) {
            const int row = e + 8 * half;
            h1[wave][row * 68 + nt * 16 + m] = fmaxf(c[e], 0.f);
        }
    }

    // ---- build layer-2 A fragments from LDS (layout change D -> A) ----
    bf16x16 a2[2];
    const float* myrow = &h1[wave][m * 68];
#pragma unroll
    for (int kc = 0; kc < 2; ++kc) {
        const int k0 = 32 * kc + 8 * half;
        const int k1 = k0 + 16;
#pragma unroll
        for (int j = 0; j < 8; j += 2) {      // pairwise -> packed bf16 converts
            float lo0 = myrow[k0 + j], lo1 = myrow[k0 + j + 1];
            float hi0 = myrow[k1 + j], hi1 = myrow[k1 + j + 1];
            a2[kc][j]         = f2bf(lo0);
            a2[kc][j + 1]     = f2bf(lo1);
            a2[kc][8 + j]     = f2bf(hi0);
            a2[kc][8 + j + 1] = f2bf(hi1);
        }
    }

    // ---- layer 2 + head ----
    float t[8];
#pragma unroll
    for (int e = 0; e < 8; ++e) t[e] = 0.f;
#pragma unroll
    for (int nt = 0; nt < 4; ++nt) {
        const float bias = b2[nt * 16 + m];
        floatx8 c;
#pragma unroll
        for (int e = 0; e < 8; ++e) c[e] = bias;
#pragma unroll
        for (int kc = 0; kc < 2; ++kc) {
            const bf16x16 bw =
                *(const bf16x16*)(w2f + (size_t)((nt * 2 + kc) * 32 + lane) * 16);
            c = __builtin_amdgcn_wmma_f32_16x16x32_bf16(
                    false, a2[kc], false, bw, (short)0, c, false, false);
        }
        const float w3n = W3[nt * 16 + m];    // column n = lane&15 of this N-tile
#pragma unroll
        for (int e = 0; e < 8; ++e) t[e] += fmaxf(c[e], 0.f) * w3n;
    }
    // reduce over the 16 columns held across each 16-lane half-group
#pragma unroll
    for (int mask = 8; mask >= 1; mask >>= 1)
#pragma unroll
        for (int e = 0; e < 8; ++e) t[e] += __shfl_xor(t[e], mask, 32);

    if (m == 0) {                             // lanes 0 (rows 0..7) and 16 (rows 8..15)
        const float bb = b3[0];
        float* out = scores + tile * 16 + 8 * half;
#pragma unroll
        for (int e = 0; e < 8; ++e) out[e] = t[e] + bb;
    }
}

// ---------------------------------------------------------------------
// Kernel 3: segment max (atomic on monotone uint key).
// ---------------------------------------------------------------------
__global__ void segmax_kernel(const float* __restrict__ scores,
                              const int* __restrict__ dst,
                              unsigned* __restrict__ maxkey, int E) {
    int e = blockIdx.x * blockDim.x + threadIdx.x;
    if (e >= E) return;
    atomicMax(&maxkey[dst[e]], flipf(scores[e]));
}

// ---------------------------------------------------------------------
// Kernel 4: e = exp(score - segmax[dst]); segment sum. (in-place over scores)
// ---------------------------------------------------------------------
__global__ void expsum_kernel(float* __restrict__ scores,
                              const int* __restrict__ dst,
                              const unsigned* __restrict__ maxkey,
                              float* __restrict__ segsum, int E) {
    int e = blockIdx.x * blockDim.x + threadIdx.x;
    if (e >= E) return;
    const int d = dst[e];
    const float ev = __expf(scores[e] - unflipf(maxkey[d]));
    scores[e] = ev;
    atomicAdd(&segsum[d], ev);
}

// ---------------------------------------------------------------------
// Kernel 5: weighted message + scatter-sum:  agg[dst] += embed_u * att
// one thread per (edge, dim)
// ---------------------------------------------------------------------
__global__ void aggregate_kernel(const float* __restrict__ embed_u,
                                 const float* __restrict__ escore,
                                 const int* __restrict__ dst,
                                 const float* __restrict__ segsum,
                                 float* __restrict__ agg, int E) {
    int t = blockIdx.x * blockDim.x + threadIdx.x;
    if (t >= E * DIMS) return;
    const int e = t >> 6;
    const int dim = t & (DIMS - 1);
    const int d = dst[e];
    const float att = escore[e] / segsum[d];
    atomicAdd(&agg[(size_t)d * DIMS + dim], embed_u[(size_t)e * DIMS + dim] * att);
}

// ---------------------------------------------------------------------
// Kernel 6: gather requested nodes.
// ---------------------------------------------------------------------
__global__ void gather_kernel(const float* __restrict__ agg,
                              const int* __restrict__ nodes,
                              float* __restrict__ out, int B) {
    int t = blockIdx.x * blockDim.x + threadIdx.x;
    if (t >= B * DIMS) return;
    const int b = t >> 6;
    const int dim = t & (DIMS - 1);
    out[t] = agg[(size_t)nodes[b] * DIMS + dim];
}

// ---------------------------------------------------------------------
static inline size_t align256(size_t x) { return (x + 255) & ~(size_t)255; }

extern "C" void kernel_launch(void* const* d_in, const int* in_sizes, int n_in,
                              void* d_out, int out_size, void* d_ws, size_t ws_size,
                              hipStream_t stream) {
    const float* embed_u = (const float*)d_in[0];
    const float* rep     = (const float*)d_in[1];
    const int*   dst     = (const int*)d_in[2];
    const int*   nodes   = (const int*)d_in[3];
    const float* W1      = (const float*)d_in[4];
    const float* b1      = (const float*)d_in[5];
    const float* W2      = (const float*)d_in[6];
    const float* b2      = (const float*)d_in[7];
    const float* W3      = (const float*)d_in[8];
    const float* b3      = (const float*)d_in[9];
    float* out = (float*)d_out;

    const int E = in_sizes[2];   // 1,000,000
    const int B = in_sizes[3];   // 16,384

    // ---- workspace carve-up ----
    char* ws = (char*)d_ws;
    size_t off = 0;
    __bf16* w1f = (__bf16*)(ws + off); off = align256(off + 4 * 4 * 32 * 16 * sizeof(__bf16));
    __bf16* w2f = (__bf16*)(ws + off); off = align256(off + 4 * 2 * 32 * 16 * sizeof(__bf16));
    float* scores    = (float*)(ws + off);    off = align256(off + (size_t)E * sizeof(float));
    unsigned* maxkey = (unsigned*)(ws + off); off = align256(off + (size_t)N_NODES * sizeof(unsigned));
    float* segsum    = (float*)(ws + off);    off = align256(off + (size_t)N_NODES * sizeof(float));
    float* agg       = (float*)(ws + off);    off = align256(off + (size_t)N_NODES * DIMS * sizeof(float));
    (void)ws_size;

    // 0) weight swizzle into WMMA B-fragment layout
    prep_weights_kernel<<<32, 256, 0, stream>>>(W1, W2, w1f, w2f);

    // 1) init segment buffers
    init_segments_kernel<<<(N_NODES * DIMS + 255) / 256, 256, 0, stream>>>(maxkey, segsum, agg);

    // 2) edge MLP scores via WMMA: 16 edges/wave, 4 waves/block
    const int tiles  = (E + 15) / 16;
    const int blocks = (tiles + 3) / 4;
    mlp_score_kernel<<<blocks, 128, 0, stream>>>(embed_u, rep, w1f, w2f,
                                                 b1, b2, W3, b3, scores, E);

    // 3) edge softmax over destination segments
    segmax_kernel<<<(E + 255) / 256, 256, 0, stream>>>(scores, dst, maxkey, E);
    expsum_kernel<<<(E + 255) / 256, 256, 0, stream>>>(scores, dst, maxkey, segsum, E);

    // 4) weighted scatter aggregation
    aggregate_kernel<<<((size_t)E * DIMS + 255) / 256, 256, 0, stream>>>(
        embed_u, scores, dst, segsum, agg, E);

    // 5) gather output rows
    gather_kernel<<<(B * DIMS + 255) / 256, 256, 0, stream>>>(agg, nodes, out, B);
}
